// XiRNNBase_34119220199571
// MI455X (gfx1250) — matrix operand order repaired and checked by reading, not compiled, not run on hardware
//
#include <hip/hip_runtime.h>
#include <math.h>

// Problem constants (match reference)
#define kB 64
#define kT 188
#define kI 512
#define kH 1024
#define kR 1024
#define kG 4096   // 4*H

typedef __attribute__((ext_vector_type(16))) __bf16 v16bf;
typedef __attribute__((ext_vector_type(8)))  float  v8f;
typedef int v4i __attribute__((vector_size(4 * sizeof(int))));

struct __align__(16) BF8 { __bf16 e[8]; };
union AFr { v16bf v; BF8 h[2]; };

#define AS1 __attribute__((address_space(1)))
#define AS3 __attribute__((address_space(3)))

#if defined(__has_builtin)
#if __has_builtin(__builtin_amdgcn_global_load_async_to_lds_b128)
#define USE_ASYNC_LDS 1
#endif
#endif
#ifndef USE_ASYNC_LDS
#define USE_ASYNC_LDS 0
#endif

__device__ __forceinline__ void wait_asynccnt0() {
#if defined(__has_builtin) && __has_builtin(__builtin_amdgcn_s_wait_asynccnt)
  __builtin_amdgcn_s_wait_asynccnt(0);
#else
  asm volatile("s_wait_asynccnt 0x0" ::: "memory");
#endif
}

// ---- fragment loaders (layouts per CDNA5 ISA 7.12.2) -----------------------
// A (MxK bf16, 16x32): lane l holds row M=l%16; half=l/16:
//   vgpr0..3 = K[half*8 .. half*8+7], vgpr4..7 = K[16+half*8 .. 16+half*8+7]
__device__ __forceinline__ v16bf load_a_frag_g(const __bf16* __restrict__ row,
                                               int k0, int hf) {
  AFr u;
  u.h[0] = *(const BF8*)(row + k0 + hf * 8);
  u.h[1] = *(const BF8*)(row + k0 + 16 + hf * 8);
  return u.v;
}
// B (KxN bf16, 32x16): lane l holds column N=l%16; half=l/16:
//   K[half*8 .. half*8+7], then K[16+half*8 .. 16+half*8+7] (same as A)
__device__ __forceinline__ v16bf load_b_frag_g(const __bf16* __restrict__ wrow,
                                               int k0, int hf) {
  AFr u;
  const __bf16* p = wrow + k0 + hf * 8;
  u.h[0] = *(const BF8*)(p);
  u.h[1] = *(const BF8*)(p + 16);
  return u.v;
}

__device__ __forceinline__ v8f wmma_bf16(v16bf a, v16bf b, v8f c) {
  return __builtin_amdgcn_wmma_f32_16x16x32_bf16(
      /*neg_a=*/false, a, /*neg_b=*/false, b,
      /*c_mod=*/(short)0, c, /*reuse_a=*/false, /*reuse_b=*/false);
}

__device__ __forceinline__ float sigmf(float x) {
  return 1.0f / (1.0f + __expf(-x));
}

// ---- elementwise conversion / init ----------------------------------------
__global__ void f32_to_bf16_kernel(const float* __restrict__ s,
                                   __bf16* __restrict__ d, int n) {
  int i = blockIdx.x * blockDim.x + threadIdx.x;
  int stride = gridDim.x * blockDim.x;
  for (; i < n; i += stride) d[i] = (__bf16)s[i];
}

__global__ void init_state_kernel(const float* __restrict__ h0,
                                  const float* __restrict__ c0,
                                  __bf16* __restrict__ hb,
                                  __bf16* __restrict__ cb,
                                  float* __restrict__ cf, int n) {
  int i = blockIdx.x * blockDim.x + threadIdx.x;
  int stride = gridDim.x * blockDim.x;
  for (; i < n; i += stride) {
    float h = h0[i], c = c0[i];
    hb[i] = (__bf16)h;
    cb[i] = (__bf16)c;
    cf[i] = c;
  }
}

// ---- phase 1: pre_x[b*T+t, g] = sum_i x[b,t,i] * W_ih[g,i] + bias[g] ------
// grid = (M/64, N/64), block = 128 (4 waves). Wave w: rows [blk.x*64+16w, +16),
// cols [blk.y*64, +64) as four 16x16 accumulators. Straight-from-L2 operands;
// disasm shows this pipelines well (clauses + partial loadcnt waits).
__global__ void __launch_bounds__(128)
prex_gemm_kernel(const __bf16* __restrict__ xb,    // [B*T, I]
                 const __bf16* __restrict__ wih,   // [4H, I]
                 const float*  __restrict__ bias,  // [4H]
                 float* __restrict__ prex) {       // [B*T, 4H]
  const int lane = threadIdx.x & 31;
  const int wave = threadIdx.x >> 5;
  const int hf   = lane >> 4;
  const int l16  = lane & 15;
  const int m0 = blockIdx.x * 64 + wave * 16;
  const int n0 = blockIdx.y * 64;

  const __bf16* arow = xb + (size_t)(m0 + l16) * kI;
  const __bf16* b0r = wih + (size_t)(n0 + 0  + l16) * kI;
  const __bf16* b1r = wih + (size_t)(n0 + 16 + l16) * kI;
  const __bf16* b2r = wih + (size_t)(n0 + 32 + l16) * kI;
  const __bf16* b3r = wih + (size_t)(n0 + 48 + l16) * kI;

  v8f a0 = {}, a1 = {}, a2 = {}, a3 = {};
#pragma unroll 4
  for (int k0 = 0; k0 < kI; k0 += 32) {
    v16bf af = load_a_frag_g(arow, k0, hf);
    a0 = wmma_bf16(af, load_b_frag_g(b0r, k0, hf), a0);
    a1 = wmma_bf16(af, load_b_frag_g(b1r, k0, hf), a1);
    a2 = wmma_bf16(af, load_b_frag_g(b2r, k0, hf), a2);
    a3 = wmma_bf16(af, load_b_frag_g(b3r, k0, hf), a3);
  }
  const int c = n0 + l16;
  const float bi0 = bias[c], bi1 = bias[c + 16], bi2 = bias[c + 32],
              bi3 = bias[c + 48];
#pragma unroll
  for (int r = 0; r < 8; r++) {
    size_t row = (size_t)(m0 + r + 8 * hf) * kG;
    prex[row + c]      = a0[r] + bi0;
    prex[row + c + 16] = a1[r] + bi1;
    prex[row + c + 32] = a2[r] + bi2;
    prex[row + c + 48] = a3[r] + bi3;
  }
}

// ---- phase 2: one recurrent LSTM step with LDS-staged operands -------------
// grid = H/16 = 64 blocks, block = 128 (4 waves). Block owns hidden units
// j in [16*blk, +16); wave w owns batch rows [16w, +16).
//
// Per k-step the block needs: h slice 64x32 (4KB), c slice 64x32 (4KB),
// 6 weight tiles 16x32 (6KB). Staged into LDS once per block, double
// buffered. Preferred producer: GLOBAL_LOAD_ASYNC_TO_LDS_B128 (ASYNCcnt,
// one barrier per k-step); fallback: register prefetch + ds_store.
// Rows padded 32->40 bf16 so the 16 rows a lane-group reads start on
// distinct bank groups (40 elems = 20 dwords; 16*20 spans all 64 banks).

#define ROWP 40                      // padded row length (bf16 elems)
#define SH_H 0                       // h slice:   64 rows
#define SH_C (64 * ROWP)             // c slice:   64 rows
#define SH_B (128 * ROWP)            // 6 weight tiles of 16 rows
#define BUFSZ (128 * ROWP + 6 * 16 * ROWP)   // 8960 bf16 per buffer
#define NCHUNK 896                   // 16B chunks per slice (= 128 thr * 7)

__device__ __forceinline__ const __bf16* chunk_src(
    int idx, int k0, int j0, const __bf16* __restrict__ hb,
    const __bf16* __restrict__ cbuf, const __bf16* __restrict__ whh,
    const __bf16* __restrict__ wrec) {
  if (idx < 256) {                     // h slice: 64 rows x 4 chunks
    int row = idx >> 2, cl = idx & 3;
    return hb + (size_t)row * kR + k0 + cl * 8;
  }
  if (idx < 512) {                     // c slice
    int i2 = idx - 256;
    int row = i2 >> 2, cl = i2 & 3;
    return cbuf + (size_t)row * kH + k0 + cl * 8;
  }
  int i2 = idx - 512;                  // weight tiles: g in [0,6)
  int g = i2 >> 6, row = (i2 >> 2) & 15, cl = i2 & 3;
  if (g < 4) return whh + (size_t)(g * kH + j0 + row) * kR + k0 + cl * 8;
  return wrec + (size_t)((g - 4) * kR + j0 + row) * kH + k0 + cl * 8;
}

__device__ __forceinline__ int chunk_dst(int idx) {  // offset in bf16 elems
  if (idx < 256) {
    int row = idx >> 2, cl = idx & 3;
    return SH_H + row * ROWP + cl * 8;
  }
  if (idx < 512) {
    int i2 = idx - 256;
    return SH_C + (i2 >> 2) * ROWP + (i2 & 3) * 8;
  }
  int i2 = idx - 512;
  int g = i2 >> 6, row = (i2 >> 2) & 15, cl = i2 & 3;
  return SH_B + g * (16 * ROWP) + row * ROWP + cl * 8;
}

#if USE_ASYNC_LDS
// Issue the 7 async 16B global->LDS copies this thread owns for one k-slice.
__device__ __forceinline__ void stage_slice_async(
    __bf16* bufbase, int tid, int k0, int j0,
    const __bf16* __restrict__ hb, const __bf16* __restrict__ cbuf,
    const __bf16* __restrict__ whh, const __bf16* __restrict__ wrec) {
#pragma unroll
  for (int q = 0; q < 7; q++) {
    const int idx = tid + q * 128;
    const __bf16* src = chunk_src(idx, k0, j0, hb, cbuf, whh, wrec);
    __bf16* dst = bufbase + chunk_dst(idx);
    // flat LDS address low 32 bits == LDS byte offset (aperture rule)
    __builtin_amdgcn_global_load_async_to_lds_b128(
        (AS1 v4i*)(uintptr_t)src,
        (AS3 v4i*)(unsigned)(uintptr_t)dst, 0, 0);
  }
}
#endif

__device__ __forceinline__ v16bf lds_a_frag(const __bf16* base, int row,
                                            int hf) {
  AFr u;
  const __bf16* p = base + row * ROWP;
  u.h[0] = *(const BF8*)(p + hf * 8);
  u.h[1] = *(const BF8*)(p + 16 + hf * 8);
  return u.v;
}
__device__ __forceinline__ v16bf lds_b_frag(const __bf16* tile, int l16,
                                            int hf) {
  AFr u;
  const __bf16* p = tile + l16 * ROWP + hf * 8;
  u.h[0] = *(const BF8*)(p);
  u.h[1] = *(const BF8*)(p + 16);
  return u.v;
}

__global__ void __launch_bounds__(128)
lstm_step_kernel(const float*  __restrict__ prex,   // [B*T, 4H]
                 const __bf16* __restrict__ whh,    // [4H, R] bf16
                 const __bf16* __restrict__ wrec,   // [2R, H] bf16
                 const __bf16* __restrict__ hb_in,  // [B, R]
                 const __bf16* __restrict__ cb_in,  // [B, H]
                 const float*  __restrict__ cf_in,  // [B, H]
                 __bf16* __restrict__ hb_out,
                 __bf16* __restrict__ cb_out,
                 float*  __restrict__ cf_out,
                 const int* __restrict__ lengths,   // [B]
                 float* __restrict__ hs,            // [B, T, H]
                 float* __restrict__ cs,            // [B, T, H]
                 int t) {
  __shared__ __bf16 smem[2 * BUFSZ];

  const int tid  = threadIdx.x;
  const int lane = tid & 31;
  const int wave = tid >> 5;
  const int hf   = lane >> 4;
  const int l16  = lane & 15;
  const int m0 = wave * 16;           // batch tile
  const int j0 = blockIdx.x * 16;     // hidden-unit tile

  v8f Ai = {}, Af = {}, Ag = {}, Ao = {}, Pi = {}, Pf = {};

#if USE_ASYNC_LDS
  // ---- async producer pipeline: one barrier per k-step ----
  stage_slice_async(smem, tid, 0, j0, hb_in, cb_in, whh, wrec);
#pragma unroll 2
  for (int ks = 0; ks < kR / 32; ks++) {
    wait_asynccnt0();     // slice ks resident in LDS (this wave's copies)
    __syncthreads();      // all waves' copies visible; all done reading other
                          // buffer from iteration ks-1
    if (ks + 1 < kR / 32)
      stage_slice_async(smem + ((ks + 1) & 1) * BUFSZ, tid, (ks + 1) * 32, j0,
                        hb_in, cb_in, whh, wrec);

    const __bf16* buf = smem + (ks & 1) * BUFSZ;
    v16bf afh = lds_a_frag(buf + SH_H, m0 + l16, hf);
    v16bf afc = lds_a_frag(buf + SH_C, m0 + l16, hf);
    Ai = wmma_bf16(afh, lds_b_frag(buf + SH_B + 0 * 16 * ROWP, l16, hf), Ai);
    Af = wmma_bf16(afh, lds_b_frag(buf + SH_B + 1 * 16 * ROWP, l16, hf), Af);
    Ag = wmma_bf16(afh, lds_b_frag(buf + SH_B + 2 * 16 * ROWP, l16, hf), Ag);
    Ao = wmma_bf16(afh, lds_b_frag(buf + SH_B + 3 * 16 * ROWP, l16, hf), Ao);
    Pi = wmma_bf16(afc, lds_b_frag(buf + SH_B + 4 * 16 * ROWP, l16, hf), Pi);
    Pf = wmma_bf16(afc, lds_b_frag(buf + SH_B + 5 * 16 * ROWP, l16, hf), Pf);
  }
#else
  // ---- fallback: register prefetch + ds_store double buffer ----
  BF8 pf[7];
#pragma unroll
  for (int q = 0; q < 7; q++)
    pf[q] = *(const BF8*)chunk_src(tid + q * 128, 0, j0, hb_in, cb_in, whh,
                                   wrec);
#pragma unroll
  for (int q = 0; q < 7; q++)
    *(BF8*)(smem + chunk_dst(tid + q * 128)) = pf[q];
  __syncthreads();

#pragma unroll 2
  for (int ks = 0; ks < kR / 32; ks++) {
    const __bf16* buf = smem + (ks & 1) * BUFSZ;

    if (ks + 1 < kR / 32) {
      const int k0n = (ks + 1) * 32;
#pragma unroll
      for (int q = 0; q < 7; q++)
        pf[q] = *(const BF8*)chunk_src(tid + q * 128, k0n, j0, hb_in, cb_in,
                                       whh, wrec);
    }

    v16bf afh = lds_a_frag(buf + SH_H, m0 + l16, hf);
    v16bf afc = lds_a_frag(buf + SH_C, m0 + l16, hf);
    Ai = wmma_bf16(afh, lds_b_frag(buf + SH_B + 0 * 16 * ROWP, l16, hf), Ai);
    Af = wmma_bf16(afh, lds_b_frag(buf + SH_B + 1 * 16 * ROWP, l16, hf), Af);
    Ag = wmma_bf16(afh, lds_b_frag(buf + SH_B + 2 * 16 * ROWP, l16, hf), Ag);
    Ao = wmma_bf16(afh, lds_b_frag(buf + SH_B + 3 * 16 * ROWP, l16, hf), Ao);
    Pi = wmma_bf16(afc, lds_b_frag(buf + SH_B + 4 * 16 * ROWP, l16, hf), Pi);
    Pf = wmma_bf16(afc, lds_b_frag(buf + SH_B + 5 * 16 * ROWP, l16, hf), Pf);

    if (ks + 1 < kR / 32) {
      __syncthreads();
      __bf16* nbuf = smem + ((ks + 1) & 1) * BUFSZ;
#pragma unroll
      for (int q = 0; q < 7; q++)
        *(BF8*)(nbuf + chunk_dst(tid + q * 128)) = pf[q];
      __syncthreads();
    }
  }
#endif

  // ---- epilogue: gates, state update, masking, stores ----
  const int j = j0 + l16;
#pragma unroll
  for (int r = 0; r < 8; r++) {
    const int b = m0 + r + 8 * hf;
    const size_t pxrow = ((size_t)b * kT + t) * kG;
    const float px_i = prex[pxrow + 0 * kH + j];
    const float px_f = prex[pxrow + 1 * kH + j];
    const float px_g = prex[pxrow + 2 * kH + j];
    const float px_o = prex[pxrow + 3 * kH + j];
    const float c_old = cf_in[(size_t)b * kH + j];

    const float iv = sigmf(Ai[r] + px_i + Pi[r]);
    const float fv = sigmf(Af[r] + px_f + Pf[r]);
    const float gv = tanhf(Ag[r] + px_g);
    float cn = fv * c_old + iv * gv;
    const float ov = sigmf(Ao[r] + px_o + cn);
    float hn = ov * tanhf(cn);

    const float m = (t < lengths[b]) ? 1.0f : 0.0f;  // zero finished seqs
    hn *= m;
    cn *= m;

    const size_t oo = ((size_t)b * kT + t) * kH + j;
    hs[oo] = hn;
    cs[oo] = cn;
    hb_out[(size_t)b * kR + j] = (__bf16)hn;
    cb_out[(size_t)b * kH + j] = (__bf16)cn;
    cf_out[(size_t)b * kH + j] = cn;
  }
}

// ---- host orchestration ----------------------------------------------------
extern "C" void kernel_launch(void* const* d_in, const int* in_sizes, int n_in,
                              void* d_out, int out_size, void* d_ws,
                              size_t ws_size, hipStream_t stream) {
  const float* x      = (const float*)d_in[0];  // [B,T,I]
  const float* h0     = (const float*)d_in[1];  // [B,R]
  const float* c0     = (const float*)d_in[2];  // [B,H]
  const float* wih    = (const float*)d_in[3];  // [4H,I]
  const float* whh    = (const float*)d_in[4];  // [4H,R]
  const float* wrec   = (const float*)d_in[5];  // [2R,H]
  const float* bias   = (const float*)d_in[6];  // [4H]
  const int*   lens   = (const int*)d_in[7];    // [B]

  char* ws = (char*)d_ws;
  size_t off = 0;
  auto take = [&](size_t bytes) -> char* {
    char* p = ws + off;
    off = (off + bytes + 255) & ~(size_t)255;
    return p;
  };

  __bf16* xb    = (__bf16*)take((size_t)kB * kT * kI * 2);
  __bf16* wihb  = (__bf16*)take((size_t)kG * kI * 2);
  __bf16* whhb  = (__bf16*)take((size_t)kG * kR * 2);
  __bf16* wrecb = (__bf16*)take((size_t)2 * kR * kH * 2);
  float*  prex  = (float*)take((size_t)kB * kT * kG * 4);
  __bf16* hb[2] = {(__bf16*)take((size_t)kB * kR * 2),
                   (__bf16*)take((size_t)kB * kR * 2)};
  __bf16* cb[2] = {(__bf16*)take((size_t)kB * kH * 2),
                   (__bf16*)take((size_t)kB * kH * 2)};
  float*  cf[2] = {(float*)take((size_t)kB * kH * 4),
                   (float*)take((size_t)kB * kH * 4)};

  // bf16 conversions
  {
    int n;
    n = kB * kT * kI;
    f32_to_bf16_kernel<<<(n + 255) / 256, 256, 0, stream>>>(x, xb, n);
    n = kG * kI;
    f32_to_bf16_kernel<<<(n + 255) / 256, 256, 0, stream>>>(wih, wihb, n);
    n = kG * kR;
    f32_to_bf16_kernel<<<(n + 255) / 256, 256, 0, stream>>>(whh, whhb, n);
    n = 2 * kR * kH;
    f32_to_bf16_kernel<<<(n + 255) / 256, 256, 0, stream>>>(wrec, wrecb, n);
    n = kB * kH;
    init_state_kernel<<<(n + 255) / 256, 256, 0, stream>>>(h0, c0, hb[0],
                                                           cb[0], cf[0], n);
  }

  // phase 1: pre_x GEMM (M = B*T = 12032, N = 4H = 4096, K = I = 512)
  {
    dim3 grid((kB * kT) / 64, kG / 64);
    prex_gemm_kernel<<<grid, 128, 0, stream>>>(xb, wihb, bias, prex);
  }

  // phase 2: sequential scan, one launch per timestep (graph-capture safe)
  float* hs = (float*)d_out;
  float* cs = hs + (size_t)kB * kT * kH;
  for (int t = 0; t < kT; t++) {
    const int cur = t & 1, nxt = cur ^ 1;
    lstm_step_kernel<<<kH / 16, 128, 0, stream>>>(
        prex, whhb, wrecb, hb[cur], cb[cur], cf[cur], hb[nxt], cb[nxt],
        cf[nxt], lens, hs, cs, t);
  }
}
